// WindowedAttention_249108103230
// MI455X (gfx1250) — compile-verified, matrix operand order checked by reading, not run on hardware
//
#include <hip/hip_runtime.h>

#define DEV __device__ __forceinline__

typedef __bf16 v16bf __attribute__((ext_vector_type(16)));
typedef float  v8f   __attribute__((ext_vector_type(8)));
typedef float  v4f   __attribute__((ext_vector_type(4)));
typedef unsigned int v4u __attribute__((ext_vector_type(4)));

union FragBF {
    v16bf v;
    __bf16 h[16];
    v4u u[2];
};

// Per-lane fragment load from a row-major bf16 matrix (A 16x32 / B 32x16).
// p points at (row, k_base); runs [0..7] and [16..23] per CDNA5 ISA 7.12.2.
DEV v16bf load_frag(const __bf16* p) {
    FragBF f;
    f.u[0] = *(const v4u*)(p);
    f.u[1] = *(const v4u*)(p + 16);
    return f.v;
}

DEV v8f wmma_bf16(v16bf a, v16bf b, v8f c) {
    return __builtin_amdgcn_wmma_f32_16x16x32_bf16(
        /*neg_a=*/false, a, /*neg_b=*/false, b,
        /*c_mod=*/(short)0, c, /*reuse_a=*/false, /*reuse_b=*/false);
}

// Async global->LDS copy, 16 bytes per lane (ASYNCcnt-tracked, no VGPRs).
// lds_off is a byte offset from the wave's LDS allocation base (our single
// __shared__ block starts at offset 0).
DEV void async_ld_b128(unsigned lds_off, const void* gaddr) {
    asm volatile("global_load_async_to_lds_b128 %0, %1, off"
                 :: "v"(lds_off), "v"(gaddr) : "memory");
}

template <int N> DEV void wait_async() {
#if __has_builtin(__builtin_amdgcn_s_wait_asynccnt)
    __builtin_amdgcn_s_wait_asynccnt((unsigned short)N);
#else
    asm volatile("s_wait_asynccnt %0" :: "i"(N) : "memory");
#endif
}

// ---------------------------------------------------------------------------
// fp32 -> bf16 elementwise convert (bandwidth-bound prepass).
// ---------------------------------------------------------------------------
__global__ __launch_bounds__(256) void cvt_bf16_kernel(
    const float* __restrict__ in, __bf16* __restrict__ out, int n4)
{
    const int i = blockIdx.x * 256 + threadIdx.x;
    if (i < n4) {
        v4f x = *(const v4f*)(in + (size_t)i * 4);
        union { __bf16 h[4]; unsigned long long u; } p;
#pragma unroll
        for (int j = 0; j < 4; ++j) p.h[j] = (__bf16)x[j];
        *(unsigned long long*)(out + (size_t)i * 4) = p.u;
    }
}

// ---------------------------------------------------------------------------
// GEMM: Y[M,N] = X[M,K] * W[N,K]^T + bias, bf16 WMMA, fp32 accumulate.
// Block: 256 threads = 8 waves (2M x 4N); wave tile 32x64; block tile 64x256.
// A/B k-slices double-buffered in LDS via global_load_async_to_lds_b128.
// ---------------------------------------------------------------------------
template <typename TOUT>
__global__ __launch_bounds__(256) void gemm_bias_kernel(
    const __bf16* __restrict__ X, const __bf16* __restrict__ W,
    const float* __restrict__ bias, TOUT* __restrict__ Y,
    int M, int N, int K)
{
    // [0, 8K): As[2][64*32] bf16; [8K, 40K): Bs[2][256*32] bf16
    __shared__ __align__(16) char smem[40960];

    const int tid  = threadIdx.x;
    const int lane = tid & 31;
    const int wave = tid >> 5;
    const int wm   = wave >> 2;   // 0..1
    const int wn   = wave & 3;    // 0..3
    const int l15  = lane & 15;
    const int half = lane >> 4;

    const int mblk = blockIdx.y * 64;
    const int nblk = blockIdx.x * 256;
    const int m0   = mblk + wm * 32;
    const int n0   = nblk + wn * 64;

    v8f acc[2][4];
#pragma unroll
    for (int t = 0; t < 2; ++t)
#pragma unroll
        for (int j = 0; j < 4; ++j)
#pragma unroll
            for (int i = 0; i < 8; ++i) acc[t][j][i] = 0.f;

    // Stage one 32-wide k-slice of A (64x32) and B (256x32) into buffer `buf`.
    // Per wave: 1 A-chunk + 4 B-chunks = 5 async instructions.
    auto stage = [&](int buf, int k0) {
        {   // A tile: 256 x 16B chunks, one per thread
            const int row = tid >> 2, c8 = (tid & 3) * 8;
            const unsigned lofs = (unsigned)(buf * 4096 + row * 64 + c8 * 2);
            async_ld_b128(lofs, X + (size_t)(mblk + row) * K + k0 + c8);
        }
#pragma unroll
        for (int r = 0; r < 4; ++r) {  // B tile: 1024 chunks, 4 per thread
            const int id  = tid + r * 256;
            const int row = id >> 2, c8 = (id & 3) * 8;
            const unsigned lofs = (unsigned)(8192 + buf * 16384 + row * 64 + c8 * 2);
            async_ld_b128(lofs, W + (size_t)(nblk + row) * K + k0 + c8);
        }
    };

    const int NT = K / 32;
    stage(0, 0);

    for (int kt = 0; kt < NT; ++kt) {
        const int cur = kt & 1;
        if (kt + 1 < NT) {
            stage(cur ^ 1, (kt + 1) * 32);
            wait_async<5>();   // drain the 5 copies belonging to `cur`
        } else {
            wait_async<0>();
        }
        __syncthreads();       // cur buffer visible to all waves

        const __bf16* Ab = (const __bf16*)(smem + cur * 4096);
        const __bf16* Bb = (const __bf16*)(smem + 8192 + cur * 16384);
        v16bf a[2], b[4];
#pragma unroll
        for (int t = 0; t < 2; ++t)
            a[t] = load_frag(Ab + (wm * 32 + t * 16 + l15) * 32 + half * 8);
#pragma unroll
        for (int j = 0; j < 4; ++j)
            b[j] = load_frag(Bb + (wn * 64 + j * 16 + l15) * 32 + half * 8);
#pragma unroll
        for (int t = 0; t < 2; ++t)
#pragma unroll
            for (int j = 0; j < 4; ++j)
                acc[t][j] = wmma_bf16(a[t], b[j], acc[t][j]);

        __syncthreads();       // everyone done reading cur before re-staging it
    }

#pragma unroll
    for (int j = 0; j < 4; ++j) {
        const int col = n0 + j * 16 + l15;
        const float bb = bias[col];
#pragma unroll
        for (int t = 0; t < 2; ++t) {
#pragma unroll
            for (int i = 0; i < 8; ++i) {
                const int row = m0 + t * 16 + i + 8 * half;
                Y[(size_t)row * N + col] = (TOUT)(acc[t][j][i] + bb);
            }
        }
    }
}

// ---------------------------------------------------------------------------
// Windowed attention: one block per (b, h, window). 512 threads = 16 waves,
// each wave owns 16 query rows. Flash-style online softmax over key chunks
// of 32. V staged transposed in LDS; P round-trips a per-wave LDS tile to
// convert C-layout -> A-layout (guarded by s_wait_dscnt).
// ---------------------------------------------------------------------------
#define S_TOT 4096
#define D_TOT 1024
#define WIN_N 256
#define DK_N  64

__global__ __launch_bounds__(512) void win_attn_kernel(
    const __bf16* __restrict__ Q, const __bf16* __restrict__ Kp,
    const __bf16* __restrict__ V, __bf16* __restrict__ O)
{
    __shared__ __bf16 Vt[DK_N * WIN_N];     // 32 KB, [dk][key]
    __shared__ __bf16 Pst[16 * 16 * 32];    // 16 KB, per-wave 16x32 P tiles

    const int tid  = threadIdx.x;
    const int lane = tid & 31;
    const int wave = tid >> 5;    // 0..15
    const int l15  = lane & 15;
    const int half = lane >> 4;

    const int blk = blockIdx.x;          // ((b*16)+h)*16 + w
    const int w = blk & 15;
    const int h = (blk >> 4) & 15;
    const int b = blk >> 8;
    const size_t base = ((size_t)b * S_TOT + (size_t)w * WIN_N) * D_TOT + h * DK_N;
    const __bf16* qp = Q + base;
    const __bf16* kp = Kp + base;
    const __bf16* vp = V + base;
    __bf16* op = O + base;

    // Stage V transposed: Vt[dk][key].
    for (int idx = tid; idx < WIN_N * DK_N / 8; idx += 512) {
        const int s  = idx >> 3;
        const int c0 = (idx & 7) * 8;
        FragBF f;
        f.u[0] = *(const v4u*)(vp + (size_t)s * D_TOT + c0);
#pragma unroll
        for (int i = 0; i < 8; ++i) Vt[(c0 + i) * WIN_N + s] = f.h[i];
    }
    __syncthreads();

    // Q A-fragments for this wave's 16 rows (DK=64 -> two k-chunks of 32).
    v16bf qf[2];
#pragma unroll
    for (int kc = 0; kc < 2; ++kc)
        qf[kc] = load_frag(qp + (size_t)(wave * 16 + l15) * D_TOT + kc * 32 + half * 8);

    v8f o_acc[4];
#pragma unroll
    for (int nv = 0; nv < 4; ++nv)
#pragma unroll
        for (int i = 0; i < 8; ++i) o_acc[nv][i] = 0.f;

    float mstat[8], lstat[8];
#pragma unroll
    for (int i = 0; i < 8; ++i) { mstat[i] = -3.0e38f; lstat[i] = 0.f; }

    __bf16* pw = Pst + wave * (16 * 32);

    for (int c = 0; c < WIN_N / 32; ++c) {
        // S = (Q * Kc^T) * scale : 2 N-tiles x 2 k-steps = 4 WMMAs
        v8f sacc[2];
#pragma unroll
        for (int j = 0; j < 2; ++j)
#pragma unroll
            for (int i = 0; i < 8; ++i) sacc[j][i] = 0.f;
#pragma unroll
        for (int j = 0; j < 2; ++j)
#pragma unroll
            for (int kc = 0; kc < 2; ++kc) {
                v16bf kf = load_frag(
                    kp + (size_t)(c * 32 + j * 16 + l15) * D_TOT + kc * 32 + half * 8);
                sacc[j] = wmma_bf16(qf[kc], kf, sacc[j]);
            }
#pragma unroll
        for (int j = 0; j < 2; ++j)
#pragma unroll
            for (int i = 0; i < 8; ++i) sacc[j][i] *= 0.125f;  // 1/sqrt(64)

        // online softmax: row max over 32 cols (2 tiles + 16 lanes/half)
        float mx[8];
#pragma unroll
        for (int i = 0; i < 8; ++i) mx[i] = fmaxf(sacc[0][i], sacc[1][i]);
#pragma unroll
        for (int sft = 8; sft >= 1; sft >>= 1)
#pragma unroll
            for (int i = 0; i < 8; ++i)
                mx[i] = fmaxf(mx[i], __shfl_xor(mx[i], sft, 32));

        float cf[8];
#pragma unroll
        for (int i = 0; i < 8; ++i) {
            const float nm = fmaxf(mstat[i], mx[i]);
            cf[i] = __expf(mstat[i] - nm);
            mstat[i] = nm;
        }
#pragma unroll
        for (int j = 0; j < 2; ++j)
#pragma unroll
            for (int i = 0; i < 8; ++i)
                sacc[j][i] = __expf(sacc[j][i] - mstat[i]);

        float sm[8];
#pragma unroll
        for (int i = 0; i < 8; ++i) sm[i] = sacc[0][i] + sacc[1][i];
#pragma unroll
        for (int sft = 8; sft >= 1; sft >>= 1)
#pragma unroll
            for (int i = 0; i < 8; ++i)
                sm[i] += __shfl_xor(sm[i], sft, 32);
#pragma unroll
        for (int i = 0; i < 8; ++i) lstat[i] = lstat[i] * cf[i] + sm[i];

#pragma unroll
        for (int nv = 0; nv < 4; ++nv)
#pragma unroll
            for (int i = 0; i < 8; ++i) o_acc[nv][i] *= cf[i];

        // P (C-layout) -> per-wave LDS tile -> reload as A-fragment
#pragma unroll
        for (int j = 0; j < 2; ++j)
#pragma unroll
            for (int i = 0; i < 8; ++i)
                pw[(i + 8 * half) * 32 + j * 16 + l15] = (__bf16)sacc[j][i];
        asm volatile("s_wait_dscnt 0x0" ::: "memory");
        v16bf pf = load_frag(pw + l15 * 32 + half * 8);

        // O += P * Vc : 4 WMMAs (Vt gives contiguous-k B fragments)
#pragma unroll
        for (int nv = 0; nv < 4; ++nv) {
            v16bf vf = load_frag(&Vt[(nv * 16 + l15) * WIN_N + c * 32 + half * 8]);
            o_acc[nv] = wmma_bf16(pf, vf, o_acc[nv]);
        }
    }

    float inv[8];
#pragma unroll
    for (int i = 0; i < 8; ++i) inv[i] = 1.0f / lstat[i];
#pragma unroll
    for (int nv = 0; nv < 4; ++nv)
#pragma unroll
        for (int i = 0; i < 8; ++i) {
            const int row = wave * 16 + i + 8 * half;
            const int col = nv * 16 + l15;
            op[(size_t)row * D_TOT + col] = (__bf16)(o_acc[nv][i] * inv[i]);
        }
}

// ---------------------------------------------------------------------------
extern "C" void kernel_launch(void* const* d_in, const int* in_sizes, int n_in,
                              void* d_out, int out_size, void* d_ws, size_t ws_size,
                              hipStream_t stream) {
    (void)in_sizes; (void)n_in; (void)out_size; (void)ws_size;
    const float* q  = (const float*)d_in[0];
    const float* k  = (const float*)d_in[1];
    const float* v  = (const float*)d_in[2];
    const float* Wq = (const float*)d_in[3];
    const float* bq = (const float*)d_in[4];
    const float* Wk = (const float*)d_in[5];
    const float* bk = (const float*)d_in[6];
    const float* Wv = (const float*)d_in[7];
    const float* bv = (const float*)d_in[8];
    const float* Wo = (const float*)d_in[9];
    const float* bo = (const float*)d_in[10];

    const int M = 4 * 4096;   // B*S
    const int N = 1024;       // D
    const int K = 1024;       // D
    const size_t elems = (size_t)M * N;
    const size_t welems = (size_t)N * K;

    __bf16* qproj = (__bf16*)d_ws;
    __bf16* kproj = qproj + elems;
    __bf16* vproj = kproj + elems;
    __bf16* aout  = vproj + elems;
    __bf16* Xbf   = aout + elems;    // reused fp32->bf16 staging of X
    __bf16* Wbf   = Xbf + elems;     // reused fp32->bf16 staging of W

    dim3 ggrid(N / 256, M / 64);     // (4, 256)
    dim3 gblk(256);
    const int xn4 = (int)(elems / 4);
    const int wn4 = (int)(welems / 4);
    dim3 cgX((xn4 + 255) / 256), cgW((wn4 + 255) / 256), cblk(256);

    // Q projection
    cvt_bf16_kernel<<<cgX, cblk, 0, stream>>>(q, Xbf, xn4);
    cvt_bf16_kernel<<<cgW, cblk, 0, stream>>>(Wq, Wbf, wn4);
    gemm_bias_kernel<__bf16><<<ggrid, gblk, 0, stream>>>(Xbf, Wbf, bq, qproj, M, N, K);
    // K projection
    cvt_bf16_kernel<<<cgX, cblk, 0, stream>>>(k, Xbf, xn4);
    cvt_bf16_kernel<<<cgW, cblk, 0, stream>>>(Wk, Wbf, wn4);
    gemm_bias_kernel<__bf16><<<ggrid, gblk, 0, stream>>>(Xbf, Wbf, bk, kproj, M, N, K);
    // V projection
    cvt_bf16_kernel<<<cgX, cblk, 0, stream>>>(v, Xbf, xn4);
    cvt_bf16_kernel<<<cgW, cblk, 0, stream>>>(Wv, Wbf, wn4);
    gemm_bias_kernel<__bf16><<<ggrid, gblk, 0, stream>>>(Xbf, Wbf, bv, vproj, M, N, K);

    // B*H*NW = 4*16*16 = 1024 windows
    win_attn_kernel<<<dim3(1024), dim3(512), 0, stream>>>(qproj, kproj, vproj, aout);

    // Output projection (fp32 result)
    cvt_bf16_kernel<<<cgW, cblk, 0, stream>>>(Wo, Wbf, wn4);
    gemm_bias_kernel<float><<<ggrid, gblk, 0, stream>>>(aout, Wbf, bo, (float*)d_out, M, N, K);
}
